// LatentLayer_17532056502357
// MI455X (gfx1250) — compile-verified, matrix operand order checked by reading, not run on hardware
//
#include <hip/hip_runtime.h>
#include <hip/hip_bf16.h>
#include <math.h>

// ---------------------------------------------------------------------------
// Fused pairwise-L2 + logsumexp ("LatentLayer") for MI455X (gfx1250, wave32).
//
// Roofline: GEMM part = 2*N*M*64 ~= 34.4 GFLOP; fused input traffic ~10 MB,
// so HBM (23.3 TB/s) is irrelevant once fused -> run the contraction on the
// WMMA pipe (v_wmma_f32_16x16x32_f16) with f32 accumulation; f16 tensors
// (~5 MB) are L2-resident. Softmax is computed entirely in the log2 domain:
// v_exp_f32 / v_log_f32 are natively base-2, so folding log2(e) into the
// scale constants removes one v_mul_f32 per exponent (9 VALU/tile saved),
// leaving the exp chain to co-execute with the XDL WMMAs.
// ---------------------------------------------------------------------------

typedef _Float16 h2  __attribute__((ext_vector_type(2)));
typedef _Float16 h8  __attribute__((ext_vector_type(8)));
typedef _Float16 h16 __attribute__((ext_vector_type(16)));
typedef float    f4v __attribute__((ext_vector_type(4)));
typedef float    f8v __attribute__((ext_vector_type(8)));

#define ZDIM 64
#define NSPLIT 8
#define LOG2E 1.4426950408889634f
#define LN2   0.6931471805599453f

static __device__ __forceinline__ h16 cat8(h8 lo, h8 hi) {
    return __builtin_shufflevector(lo, hi, 0,1,2,3,4,5,6,7,8,9,10,11,12,13,14,15);
}

// Load one 16x32 f16 fragment (A or B role; both are row-major 64-vec rows).
// ISA 7.12.2: lane<16 holds K {koff..koff+7, koff+16..koff+23} with koff=0;
// lane>=16 same with koff=8.  Two global_load_b128 per fragment.
static __device__ __forceinline__ h16 load_frag(const _Float16* __restrict__ rowbase) {
    h8 lo = *reinterpret_cast<const h8*>(rowbase);
    h8 hi = *reinterpret_cast<const h8*>(rowbase + 16);
    return cat8(lo, hi);
}

// ---------------------------------------------------------------------------
// Kernel 1: prep.  One wave per row: f32 -> tanh (z only) -> f16, row norms.
// ---------------------------------------------------------------------------
__global__ void latent_prep_kernel(const float* __restrict__ z,
                                   const float* __restrict__ e,
                                   _Float16* __restrict__ mh,
                                   _Float16* __restrict__ eh,
                                   float* __restrict__ rnorm,
                                   float* __restrict__ enorm,
                                   int N, int M) {
    int wid  = (int)((blockIdx.x * blockDim.x + threadIdx.x) >> 5);
    int lane = (int)(threadIdx.x & 31);
    if (wid >= N + M) return;

    int k0 = lane * 2;
    float v0, v1;
    _Float16* dst;
    if (wid < N) {
        const float* s = z + (size_t)wid * ZDIM + k0;
        v0 = tanhf(s[0]);
        v1 = tanhf(s[1]);
        dst = mh + (size_t)wid * ZDIM + k0;
    } else {
        int r = wid - N;
        const float* s = e + (size_t)r * ZDIM + k0;
        v0 = s[0];
        v1 = s[1];
        dst = eh + (size_t)r * ZDIM + k0;
    }
    h2 p;
    p.x = (_Float16)v0;
    p.y = (_Float16)v1;
    *reinterpret_cast<h2*>(dst) = p;

    float acc = v0 * v0 + v1 * v1;
    #pragma unroll
    for (int off = 16; off > 0; off >>= 1)
        acc += __shfl_xor(acc, off, 32);

    if (lane == 0) {
        if (wid < N) rnorm[wid] = acc;
        else         enorm[wid - N] = acc;
    }
}

// ---------------------------------------------------------------------------
// Kernel 2: fused WMMA GEMM + online logsumexp (log2 domain) over the N axis.
// Each wave owns 16 M-columns (B fragments loop-invariant in VGPRs) and
// streams a 16-row slab of N per iteration: 2x v_wmma_f32_16x16x32_f16.
// Partials pmax/psum are stored in the log2 domain.
// ---------------------------------------------------------------------------
__global__ void latent_main_kernel(const _Float16* __restrict__ mh,
                                   const _Float16* __restrict__ eh,
                                   const float* __restrict__ rnorm,
                                   const float* __restrict__ enorm,
                                   const float* __restrict__ log_sigma,
                                   const float* __restrict__ log_temperature,
                                   float* __restrict__ pmax,
                                   float* __restrict__ psum,
                                   int rows_per_split) {
    const int wave   = (int)(threadIdx.x >> 5);
    const int lane   = (int)(threadIdx.x & 31);
    const int c      = lane & 15;      // column within the wave's 16-col tile
    const int hi     = lane >> 4;      // half-wave selector
    const int m_base = (int)blockIdx.x * 128 + wave * 16;
    const int m      = m_base + c;

    const float ls = log_sigma[0];
    const float lt = log_temperature[0];
    const float T  = __builtin_amdgcn_exp2f(LOG2E * lt);
    // log2-domain scale: c1 = log2(e) * alpha / T  (negative)
    const float c1 = -0.5f * LOG2E * __builtin_amdgcn_exp2f(-2.0f * LOG2E * ls) / T;
    const float c2 = -2.0f * c1;       // coefficient of the dot product
    const float en_s = c1 * enorm[m];  // per-column constant
    (void)T;

    // Loop-invariant B fragments: the wave's 16 e-rows, K=0..31 and K=32..63.
    const _Float16* ebase = eh + (size_t)m * ZDIM + hi * 8;
    const h16 b0 = load_frag(ebase);
    const h16 b1 = load_frag(ebase + 32);

    float mx = -INFINITY;   // running max, log2 domain
    float sm = 0.0f;        // running sum of 2^(s - mx)

    const int n0   = (int)blockIdx.y * rows_per_split;
    const int nend = n0 + rows_per_split;

    #pragma unroll 2
    for (int nb = n0; nb < nend; nb += 16) {
        // A fragments: 16 mean-rows [nb, nb+16), K=0..31 and K=32..63.
        const _Float16* abase = mh + (size_t)(nb + c) * ZDIM + hi * 8;
        __builtin_prefetch(abase + 16 * ZDIM, 0, 1);  // global_prefetch_b8
        h16 a0 = load_frag(abase);
        h16 a1 = load_frag(abase + 32);

        f8v acc = {0.f, 0.f, 0.f, 0.f, 0.f, 0.f, 0.f, 0.f};
        acc = __builtin_amdgcn_wmma_f32_16x16x32_f16(false, a0, false, b0,
                                                     (short)0, acc, false, false);
        acc = __builtin_amdgcn_wmma_f32_16x16x32_f16(false, a1, false, b1,
                                                     (short)0, acc, false, false);

        // C layout: this lane's 8 values are rows nb + hi*8 + i, column m.
        const float* rp = rnorm + nb + hi * 8;
        f4v r0 = *reinterpret_cast<const f4v*>(rp);
        f4v r1 = *reinterpret_cast<const f4v*>(rp + 4);

        float s[8];
        #pragma unroll
        for (int i = 0; i < 4; ++i) {
            s[i]     = fmaf(c2, acc[i],     fmaf(c1, r0[i], en_s));
            s[i + 4] = fmaf(c2, acc[i + 4], fmaf(c1, r1[i], en_s));
        }

        float vmax = s[0];
        #pragma unroll
        for (int i = 1; i < 8; ++i) vmax = fmaxf(vmax, s[i]);

        float nm = fmaxf(mx, vmax);
        float ssum = 0.0f;
        #pragma unroll
        for (int i = 0; i < 8; ++i) ssum += __builtin_amdgcn_exp2f(s[i] - nm);
        sm = fmaf(sm, __builtin_amdgcn_exp2f(mx - nm), ssum);
        mx = nm;
    }

    // Combine the two half-waves (same column, complementary rows).
    float omx = __shfl_xor(mx, 16, 32);
    float osm = __shfl_xor(sm, 16, 32);
    float g  = fmaxf(mx, omx);
    float s2 = sm * __builtin_amdgcn_exp2f(mx - g)
             + osm * __builtin_amdgcn_exp2f(omx - g);

    if (hi == 0) {
        pmax[(size_t)m * NSPLIT + blockIdx.y] = g;
        psum[(size_t)m * NSPLIT + blockIdx.y] = s2;
    }
}

// ---------------------------------------------------------------------------
// Kernel 3: merge the NSPLIT partial (max,sum) per column (log2 domain),
// convert to natural log, mean over M, add analytic constants, write scalar.
// ---------------------------------------------------------------------------
__global__ void latent_finish_kernel(const float* __restrict__ pmax,
                                     const float* __restrict__ psum,
                                     const float* __restrict__ log_sigma,
                                     const float* __restrict__ log_temperature,
                                     float* __restrict__ out,
                                     int M, int N) {
    __shared__ float red[256];
    const int tid = (int)threadIdx.x;

    float local = 0.0f;
    for (int m = tid; m < M; m += 256) {
        float gm = -INFINITY;
        #pragma unroll
        for (int j = 0; j < NSPLIT; ++j)
            gm = fmaxf(gm, pmax[(size_t)m * NSPLIT + j]);
        float S = 0.0f;
        #pragma unroll
        for (int j = 0; j < NSPLIT; ++j)
            S += psum[(size_t)m * NSPLIT + j]
               * __builtin_amdgcn_exp2f(pmax[(size_t)m * NSPLIT + j] - gm);
        // natural-log lse from log2-domain partials: ln2 * (gm + log2(S))
        local += LN2 * (gm + __builtin_amdgcn_logf(S));
    }
    red[tid] = local;
    __syncthreads();
    for (int off = 128; off > 0; off >>= 1) {
        if (tid < off) red[tid] += red[tid + off];
        __syncthreads();
    }
    if (tid == 0) {
        float ls = log_sigma[0];
        float lt = log_temperature[0];
        float T  = __builtin_amdgcn_exp2f(LOG2E * lt);
        float loss = -(T * red[0] / (float)M)
                   + 0.5f * (float)ZDIM * (2.0f * ls - 1.0f)
                   + LN2 * __builtin_amdgcn_logf((float)N);
        out[0] = loss;
    }
}

// ---------------------------------------------------------------------------
extern "C" void kernel_launch(void* const* d_in, const int* in_sizes, int n_in,
                              void* d_out, int out_size, void* d_ws, size_t ws_size,
                              hipStream_t stream) {
    const float* z   = (const float*)d_in[0];
    const float* e   = (const float*)d_in[1];
    const float* lsg = (const float*)d_in[2];
    const float* ltp = (const float*)d_in[3];
    float* out = (float*)d_out;

    const int N = in_sizes[0] / ZDIM;   // 32768
    const int M = in_sizes[1] / ZDIM;   // 8192

    // Workspace layout (256B aligned slabs)
    auto align256 = [](size_t x) { return (x + 255) & ~(size_t)255; };
    char* w = (char*)d_ws;
    size_t off = 0;
    _Float16* mh = (_Float16*)(w + off); off = align256(off + (size_t)N * ZDIM * sizeof(_Float16));
    _Float16* eh = (_Float16*)(w + off); off = align256(off + (size_t)M * ZDIM * sizeof(_Float16));
    float* rnorm = (float*)(w + off);    off = align256(off + (size_t)N * sizeof(float));
    float* enorm = (float*)(w + off);    off = align256(off + (size_t)M * sizeof(float));
    float* pmax  = (float*)(w + off);    off = align256(off + (size_t)M * NSPLIT * sizeof(float));
    float* psum  = (float*)(w + off);    off = align256(off + (size_t)M * NSPLIT * sizeof(float));
    (void)ws_size; (void)n_in; (void)out_size;

    // 1) prep: one wave per row, 8 waves per 256-thread block
    {
        int rows = N + M;
        int blocks = (rows + 7) / 8;
        latent_prep_kernel<<<blocks, 256, 0, stream>>>(z, e, mh, eh, rnorm, enorm, N, M);
    }

    // 2) fused WMMA + online logsumexp: grid (M/128 column blocks, NSPLIT row slices)
    {
        dim3 grid(M / 128, NSPLIT);
        int rows_per_split = N / NSPLIT;
        latent_main_kernel<<<grid, 256, 0, stream>>>(mh, eh, rnorm, enorm,
                                                     lsg, ltp, pmax, psum,
                                                     rows_per_split);
    }

    // 3) merge partials -> scalar loss
    latent_finish_kernel<<<1, 256, 0, stream>>>(pmax, psum, lsg, ltp, out, M, N);
}
